// Gate_68324339745448
// MI455X (gfx1250) — matrix operand order repaired and checked by reading, not compiled, not run on hardware
//
#include <hip/hip_runtime.h>

typedef __attribute__((ext_vector_type(16))) __bf16 v16bf;
typedef __attribute__((ext_vector_type(8)))  __bf16 bf16x8;
typedef __attribute__((ext_vector_type(4)))  __bf16 bf16x4;
typedef __attribute__((ext_vector_type(8)))  float  v8f;
typedef __attribute__((ext_vector_type(4)))  float  f32x4;

#define DIM          4096
#define N_EXPERTS    256
#define N_GROUPS     8
#define TOPK_GROUPS  4
#define TOPK         8
#define TOK_TILE     16
#define ROUTE_SCALE  1.0f
#define NEG_INF      (-3.402823466e38f)

#define KCHUNK       128                 // K per LDS staging round (4 WMMA steps)
#define NCHUNKS      (DIM / KCHUNK)      // 32
#define XS_PITCH     136                 // bf16 per LDS row: 128 + 8 pad (68-dword stride)

// ---- prep: convert gate weights fp32 -> bf16 once per launch (2 MB, L2-resident)
__global__ __launch_bounds__(256)
void cvt_w_kernel(const float* __restrict__ w, __bf16* __restrict__ wb) {
    const size_t i = ((size_t)blockIdx.x * blockDim.x + threadIdx.x) * 4;
    f32x4 v = *(const f32x4*)(w + i);
    bf16x4 o;
#pragma unroll
    for (int j = 0; j < 4; ++j) o[j] = (__bf16)v[j];
    *(bf16x4*)(wb + i) = o;
}

// One block = 16 tokens, 512 threads = 16 waves.
// GEMM: wave w computes 16x16 (token x expert) logits for experts [16w,16w+16)
//       with v_wmma_f32_16x16x32_bf16, A staged in LDS as bf16 (converted once
//       per block), B read as preconverted bf16 from global (L2-resident).
// Routing: wave w does fp32 softmax + group-limited top-8 for token w.
__global__ __launch_bounds__(512, 1)
void gate_fused_kernel(const float* __restrict__ x,
                       const __bf16* __restrict__ wb,
                       float* __restrict__ out_w,
                       int*   __restrict__ out_i,
                       int T)
{
    __shared__ __align__(16) __bf16 xs[2][TOK_TILE][XS_PITCH]; // ~8.5 KB
    __shared__ float logits[TOK_TILE][N_EXPERTS];              // 16 KB

    const int tid  = threadIdx.x;
    const int lane = tid & 31;
    const int wave = tid >> 5;          // 0..15 -> expert tile / token id
    const int half = lane >> 4;         // 0 or 1
    const int l15  = lane & 15;
    const int t0   = blockIdx.x * TOK_TILE;

    // Cooperative staging mapping: thread -> (row, 4-float column) of x chunk.
    const int srow = tid >> 5;          // 0..15
    const int scol = (tid & 31) * 4;    // 0..124
    const float* xstage = x + (size_t)(t0 + srow) * DIM + scol;

    // B stream: expert row (wave*16 + l15) of bf16 weights.
    const __bf16* wrow = wb + (size_t)(wave * 16 + l15) * DIM;

    // Prologue: stage chunk 0 into buffer 0.
    {
        f32x4 v = *(const f32x4*)(xstage);
        bf16x4 o;
#pragma unroll
        for (int j = 0; j < 4; ++j) o[j] = (__bf16)v[j];
        *(bf16x4*)&xs[0][srow][scol] = o;
    }
    __syncthreads();

    v8f acc = {};
    for (int c = 0; c < NCHUNKS; ++c) {
        const int buf = c & 1;
        const bool more = (c + 1) < NCHUNKS;

        // Kick off next chunk's global loads early (overlaps WMMA below).
        f32x4 nxt = {};
        if (more) nxt = *(const f32x4*)(xstage + (c + 1) * KCHUNK);

        const __bf16* xsrow = &xs[buf][l15][0];
        const __bf16* wk    = wrow + c * KCHUNK;
        __builtin_prefetch(wk + KCHUNK, 0, 0);   // global_prefetch_b8 on B stream

#pragma unroll
        for (int kk = 0; kk < KCHUNK; kk += 32) {
            // A fragment (16-bit A 16x32 layout):
            //  lanes 0-15 : elems 0-7 = K[kk..kk+7],    8-15 = K[kk+16..kk+23]
            //  lanes 16-31: elems 0-7 = K[kk+8..kk+15], 8-15 = K[kk+24..kk+31]
            bf16x8 alo = *(const bf16x8*)(xsrow + kk + half * 8);
            bf16x8 ahi = *(const bf16x8*)(xsrow + kk + 16 + half * 8);
            // B fragment (16-bit B 32x16 layout):
            //  lanes 0-15 : elems 0-7 = K[kk..kk+7],    8-15 = K[kk+16..kk+23]
            //  lanes 16-31: elems 0-7 = K[kk+8..kk+15], 8-15 = K[kk+24..kk+31]
            bf16x8 blo = *(const bf16x8*)(wk + kk + half * 8);
            bf16x8 bhi = *(const bf16x8*)(wk + kk + 16 + half * 8);

            v16bf a = __builtin_shufflevector(alo, ahi, 0, 1, 2, 3, 4, 5, 6, 7,
                                              8, 9, 10, 11, 12, 13, 14, 15);
            v16bf b = __builtin_shufflevector(blo, bhi, 0, 1, 2, 3, 4, 5, 6, 7,
                                              8, 9, 10, 11, 12, 13, 14, 15);

            acc = __builtin_amdgcn_wmma_f32_16x16x32_bf16(
                    /*neg_a=*/false, a, /*neg_b=*/false, b,
                    /*c_mod=*/(short)0, acc,
                    /*reuse_a=*/false, /*reuse_b=*/false);
        }

        // Convert + store next chunk into the other buffer.
        if (more) {
            bf16x4 o;
#pragma unroll
            for (int j = 0; j < 4; ++j) o[j] = (__bf16)nxt[j];
            *(bf16x4*)&xs[buf ^ 1][srow][scol] = o;
        }
        __syncthreads();
    }

    // Spill D tile to LDS. C/D layout: n = lane&15, m = vgpr + 8*(lane>>4).
#pragma unroll
    for (int r = 0; r < 8; ++r) {
        logits[r + half * 8][wave * 16 + l15] = acc[r];
    }
    __syncthreads();

    // ---------------- Routing phase: wave w handles token w ----------------
    // Lane l owns experts e = l + 32*j, j=0..7  => group(e) == j.
    float v[8];
#pragma unroll
    for (int j = 0; j < 8; ++j) v[j] = logits[wave][lane + 32 * j];

    // Softmax in fp32 across all 256 experts.
    float mx = v[0];
#pragma unroll
    for (int j = 1; j < 8; ++j) mx = fmaxf(mx, v[j]);
#pragma unroll
    for (int off = 16; off >= 1; off >>= 1)
        mx = fmaxf(mx, __shfl_xor(mx, off, 32));

    float s = 0.f;
#pragma unroll
    for (int j = 0; j < 8; ++j) { v[j] = __expf(v[j] - mx); s += v[j]; }
#pragma unroll
    for (int off = 16; off >= 1; off >>= 1)
        s += __shfl_xor(s, off, 32);
    const float inv = 1.0f / s;
#pragma unroll
    for (int j = 0; j < 8; ++j) v[j] *= inv;    // softmax scores

    // Group scores: cross-lane max of v[j] == max over experts in group j.
    float g[8];
#pragma unroll
    for (int j = 0; j < 8; ++j) {
        float gm = v[j];
#pragma unroll
        for (int off = 16; off >= 1; off >>= 1)
            gm = fmaxf(gm, __shfl_xor(gm, off, 32));
        g[j] = gm;
    }

    // Keep top-4 groups (rank with low-index tie-break, matching top_k).
    float mv[8];
#pragma unroll
    for (int j = 0; j < 8; ++j) {
        int rank = 0;
#pragma unroll
        for (int jj = 0; jj < 8; ++jj)
            rank += (g[jj] > g[j]) || (g[jj] == g[j] && jj < j);
        mv[j] = (rank < TOPK_GROUPS) ? v[j] : NEG_INF;
    }

    // Iterative top-8 over masked scores (global argmax, low-index tie-break).
    float bw[TOPK];
    int   bi[TOPK];
#pragma unroll
    for (int i = 0; i < TOPK; ++i) {
        float bv  = NEG_INF;
        int   bid = 0x7fffffff;
#pragma unroll
        for (int j = 0; j < 8; ++j) {
            int e = lane + 32 * j;
            if (mv[j] > bv || (mv[j] == bv && e < bid)) { bv = mv[j]; bid = e; }
        }
#pragma unroll
        for (int off = 16; off >= 1; off >>= 1) {
            float ov = __shfl_xor(bv, off, 32);
            int   oi = __shfl_xor(bid, off, 32);
            if (ov > bv || (ov == bv && oi < bid)) { bv = ov; bid = oi; }
        }
        bw[i] = bv;           // unmasked softmax score (winner never masked)
        bi[i] = bid;
        // Clear the winner in its owning lane.
        if (lane == (bid & 31)) {
            const int jj = bid >> 5;
#pragma unroll
            for (int j = 0; j < 8; ++j)
                if (j == jj) mv[j] = NEG_INF;
        }
    }

    // Lane 0 of each wave writes this token's results.
    if (lane == 0) {
        const size_t t = (size_t)(t0 + wave);
#pragma unroll
        for (int i = 0; i < TOPK; ++i) {
            out_w[t * TOPK + i] = bw[i] * ROUTE_SCALE;
            out_i[t * TOPK + i] = bi[i];
        }
    }
}

extern "C" void kernel_launch(void* const* d_in, const int* in_sizes, int n_in,
                              void* d_out, int out_size, void* d_ws, size_t ws_size,
                              hipStream_t stream) {
    const float* x = (const float*)d_in[0];     // [T, 4096] fp32
    const float* w = (const float*)d_in[1];     // [256, 4096] fp32
    const int T = in_sizes[0] / DIM;

    // Workspace: bf16 copy of the gate weight (256*4096*2 = 2 MB).
    __bf16* wb = (__bf16*)d_ws;

    // d_out: [T*8] fp32 weights, then [T*8] int32 indices, concatenated flat.
    float* out_w = (float*)d_out;
    int*   out_i = (int*)d_out + (size_t)T * TOPK;

    const int n_w = N_EXPERTS * DIM;                       // 1,048,576 elems
    cvt_w_kernel<<<n_w / (256 * 4), 256, 0, stream>>>(w, wb);
    gate_fused_kernel<<<T / TOK_TILE, 512, 0, stream>>>(x, wb, out_w, out_i, T);
}
